// HmmCell_70145405878667
// MI455X (gfx1250) — matrix operand order repaired and checked by reading, not compiled, not run on hardware
//
#include <hip/hip_runtime.h>

typedef __attribute__((ext_vector_type(16))) __bf16 bf16x16;
typedef __attribute__((ext_vector_type(8)))  __bf16 bf16x8;
typedef __attribute__((ext_vector_type(8)))  float  f32x8;
typedef __attribute__((__vector_size__(4 * sizeof(int)))) int v4i;

#define S_DIM        512
#define ROWS_PER_WG  64
#define KCHUNK       32
#define NCHUNK       (S_DIM / KCHUNK)      // 16
#define SF_STRIDE    36                    // floats per LDS sf row (32 + pad) -> 144B
#define A_STRIDE     40                    // bf16 per LDS A row (32 + pad)  -> 80B
#define EPSV         1e-16f

#define SF_BYTES     (ROWS_PER_WG * SF_STRIDE * 4)   // 9216
#define AHI_BYTES    (S_DIM * A_STRIDE * 2)          // 40960
#define BUF_BYTES    (SF_BYTES + 2 * AHI_BYTES)      // 91136 per buffer
#define RS_BYTES     (ROWS_PER_WG * 4)               // 256
#define SMEM_TOTAL   (2 * BUF_BYTES + RS_BYTES)      // 182528

#define GAS __attribute__((address_space(1)))
#define LAS __attribute__((address_space(3)))

#if defined(__has_builtin)
#if __has_builtin(__builtin_amdgcn_global_load_async_to_lds_b128)
#define USE_ASYNC 1
#endif
#endif

#ifdef USE_ASYNC
#if __has_builtin(__builtin_amdgcn_s_wait_asynccnt)
#define WAIT_ASYNC(n) __builtin_amdgcn_s_wait_asynccnt(n)
#else
#define WAIT_ASYNC(n) asm volatile("s_wait_asynccnt %0" :: "i"(n) : "memory")
#endif
#define NASY 18   /* async b128 issues per wave per chunk: 16 (A hi/lo) + 2 (sf) */
// g: any generic pointer to global memory (constness dropped), l: generic pointer to LDS
#define ASYNC_CP16(g, l) \
    __builtin_amdgcn_global_load_async_to_lds_b128( \
        (GAS v4i*)(char*)(g), (LAS v4i*)(l), 0, 0)
#endif

__device__ __forceinline__ __bf16 f2bf(float f) {
    unsigned u = __builtin_bit_cast(unsigned, f);
    unsigned r = (u + 0x7fffu + ((u >> 16) & 1u)) >> 16;   // round-to-nearest-even
    unsigned short s = (unsigned short)r;
    return __builtin_bit_cast(__bf16, s);
}
__device__ __forceinline__ float bf2f(__bf16 h) {
    unsigned short s = __builtin_bit_cast(unsigned short, h);
    unsigned u = ((unsigned)s) << 16;
    return __builtin_bit_cast(float, u);
}

// ---- one-time prep: A (512x512 f32, row major) -> transposed bf16 hi/lo [n][k] in ws ----
__global__ __launch_bounds__(256, 1) void prep_A_kernel(
    const float* __restrict__ A,
    unsigned short* __restrict__ AtHi,
    unsigned short* __restrict__ AtLo)
{
    int idx = blockIdx.x * 256 + threadIdx.x;   // 0..65535 (grid 256)
    int k  = idx >> 7;                          // 0..511
    int n0 = (idx & 127) * 4;                   // 4 consecutive columns
    float4 x = *(const float4*)&A[(size_t)k * S_DIM + n0];
    float vv[4] = {x.x, x.y, x.z, x.w};
    #pragma unroll
    for (int c = 0; c < 4; ++c) {
        float v = vv[c];
        __bf16 h = f2bf(v);
        AtHi[(size_t)(n0 + c) * S_DIM + k] = __builtin_bit_cast(unsigned short, h);
        AtLo[(size_t)(n0 + c) * S_DIM + k] =
            __builtin_bit_cast(unsigned short, f2bf(v - bf2f(h)));
    }
}

// ---- staging: ws (bf16 copy, async if available) ----
__device__ __forceinline__ void stage_ws(
    const unsigned short* AtHi, const unsigned short* AtLo,
    const float* SF, int row0, int kk, char* bufBase, int tid)
{
#ifdef USE_ASYNC
    #pragma unroll
    for (int i = 0; i < 8; ++i) {
        int idx = i * 256 + tid;             // 0..2047 : 512 rows x 4 x 16B
        int n = idx >> 2;
        int q = idx & 3;
        const char* gh = (const char*)AtHi + (size_t)n * (S_DIM * 2) + kk * 2 + q * 16;
        const char* gl = (const char*)AtLo + (size_t)n * (S_DIM * 2) + kk * 2 + q * 16;
        char* lh = bufBase + SF_BYTES + n * (A_STRIDE * 2) + q * 16;
        char* ll = bufBase + SF_BYTES + AHI_BYTES + n * (A_STRIDE * 2) + q * 16;
        ASYNC_CP16(gh, lh);
        ASYNC_CP16(gl, ll);
    }
    #pragma unroll
    for (int i = 0; i < 2; ++i) {
        int idx = i * 256 + tid;             // 0..511 : 64 rows x 8 x 16B
        int r = idx >> 3;
        int q = idx & 7;
        const char* g = (const char*)(SF + (size_t)(row0 + r) * S_DIM + kk) + q * 16;
        char* l = bufBase + r * (SF_STRIDE * 4) + q * 16;
        ASYNC_CP16(g, l);
    }
#else
    #pragma unroll
    for (int i = 0; i < 8; ++i) {
        int idx = i * 256 + tid;
        int n = idx >> 2;
        int q = idx & 3;
        uint4 h = *(const uint4*)((const char*)AtHi + (size_t)n * (S_DIM * 2) + kk * 2 + q * 16);
        uint4 l = *(const uint4*)((const char*)AtLo + (size_t)n * (S_DIM * 2) + kk * 2 + q * 16);
        *(uint4*)(bufBase + SF_BYTES + n * (A_STRIDE * 2) + q * 16) = h;
        *(uint4*)(bufBase + SF_BYTES + AHI_BYTES + n * (A_STRIDE * 2) + q * 16) = l;
    }
    #pragma unroll
    for (int i = 0; i < 2; ++i) {
        int idx = i * 256 + tid;
        int r = idx >> 3;
        int q = idx & 7;
        uint4 x = *(const uint4*)((const char*)(SF + (size_t)(row0 + r) * S_DIM + kk) + q * 16);
        *(uint4*)(bufBase + r * (SF_STRIDE * 4) + q * 16) = x;
    }
#endif
}

// ---- staging fallback: convert f32 A inline (no ws available) ----
__device__ __forceinline__ void stage_conv(
    const float* A, const float* SF, int row0, int kk, char* bufBase, int tid)
{
    float*          sfL = (float*)bufBase;
    unsigned short* aH  = (unsigned short*)(bufBase + SF_BYTES);
    unsigned short* aL  = (unsigned short*)(bufBase + SF_BYTES + AHI_BYTES);
    {
        int r  = tid >> 2;
        int c0 = (tid & 3) * 8;
        const float4* g = (const float4*)&SF[(size_t)(row0 + r) * S_DIM + kk + c0];
        float4 x0 = g[0], x1 = g[1];
        *(float4*)&sfL[r * SF_STRIDE + c0]     = x0;
        *(float4*)&sfL[r * SF_STRIDE + c0 + 4] = x1;
    }
    {
        int kr = tid >> 3;
        int n0 = (tid & 7) * 64;
        const float4* g = (const float4*)&A[(size_t)(kk + kr) * S_DIM + n0];
        #pragma unroll 4
        for (int i = 0; i < 16; ++i) {
            float4 x = g[i];
            float vv[4] = {x.x, x.y, x.z, x.w};
            int n = n0 + i * 4;
            #pragma unroll
            for (int c = 0; c < 4; ++c) {
                float v = vv[c];
                __bf16 h = f2bf(v);
                aH[(n + c) * A_STRIDE + kr] = __builtin_bit_cast(unsigned short, h);
                aL[(n + c) * A_STRIDE + kr] =
                    __builtin_bit_cast(unsigned short, f2bf(v - bf2f(h)));
            }
        }
    }
}

__global__ __launch_bounds__(256, 1) void hmm_fwd_step(
    const float* __restrict__ EMIT,            // (B, S)
    const float* __restrict__ SF,              // (B, S)
    const float* __restrict__ LL,              // (B, 1)
    const float* __restrict__ A,               // (S, S) f32 (fallback path)
    const unsigned short* __restrict__ AtHi,   // (S, S) bf16 transposed (ws path)
    const unsigned short* __restrict__ AtLo,
    float* __restrict__ out,                   // (B,S+1) || (B,S) || (B,1)
    int Btot, int useWs)
{
    extern __shared__ char smemRaw[];
    float* rowsum = (float*)(smemRaw + 2 * BUF_BYTES);

    const int tid  = threadIdx.x;
    const int wave = tid >> 5;
    const int lane = tid & 31;
    const int mi   = wave >> 1;            // 16-row subtile (0..3)
    const int nh   = wave & 1;             // 256-col half
    const int row0 = blockIdx.x * ROWS_PER_WG;

    if (tid < ROWS_PER_WG) rowsum[tid] = 0.0f;

    // prefetch the emission tile (consumed in the epilogue) -> global_prefetch_b8
    {
        int r  = tid >> 2;
        int l0 = tid & 3;
        #pragma unroll
        for (int j = 0; j < 4; ++j)
            __builtin_prefetch(&EMIT[(size_t)(row0 + r) * S_DIM + (l0 + 4 * j) * 32], 0, 1);
    }

    f32x8 acc[16];
    #pragma unroll
    for (int i = 0; i < 16; ++i)
        acc[i] = (f32x8){0.f,0.f,0.f,0.f,0.f,0.f,0.f,0.f};

    const int mrow  = lane & 15;
    const int lh    = lane >> 4;
    const int kbase = lh * 8;              // A-fragment K interleave base
    const int kb    = lh * 8;              // B-fragment K base
    const int ncol  = lane & 15;

    // prologue: stage chunk 0 into buffer 0
    if (useWs) stage_ws(AtHi, AtLo, SF, row0, 0, smemRaw, tid);
    else       stage_conv(A, SF, row0, 0, smemRaw, tid);

    for (int c = 0; c < NCHUNK; ++c) {
        char* buf = smemRaw + (size_t)(c & 1) * BUF_BYTES;

        if (c + 1 < NCHUNK) {
            char* nbuf = smemRaw + (size_t)((c + 1) & 1) * BUF_BYTES;
            if (useWs) stage_ws(AtHi, AtLo, SF, row0, (c + 1) * KCHUNK, nbuf, tid);
            else       stage_conv(A, SF, row0, (c + 1) * KCHUNK, nbuf, tid);
        }
#ifdef USE_ASYNC
        if (useWs) {
            if (c + 1 < NCHUNK) WAIT_ASYNC(NASY);   // chunk c's 18 completed (in-order)
            else                WAIT_ASYNC(0);
        }
#endif
        __syncthreads();

        const float*  sfLds = (const float*)buf;
        const __bf16* aHi   = (const __bf16*)(buf + SF_BYTES);
        const __bf16* aLo   = (const __bf16*)(buf + SF_BYTES + AHI_BYTES);

        // ---- A-operand (sf) fragments, hi/lo split in registers ----
        const float* sp = &sfLds[(mi * 16 + mrow) * SF_STRIDE + kbase];
        float4 p0 = ((const float4*)sp)[0];
        float4 p1 = ((const float4*)sp)[1];
        float4 p2 = ((const float4*)(sp + 16))[0];
        float4 p3 = ((const float4*)(sp + 16))[1];
        float va[16] = {p0.x,p0.y,p0.z,p0.w, p1.x,p1.y,p1.z,p1.w,
                        p2.x,p2.y,p2.z,p2.w, p3.x,p3.y,p3.z,p3.w};
        bf16x16 ahi, alo;
        #pragma unroll
        for (int j = 0; j < 16; ++j) {
            __bf16 h = f2bf(va[j]);
            ahi[j] = h;
            alo[j] = f2bf(va[j] - bf2f(h));
        }

        // ---- 16 N-tiles x 3 WMMAs (hi*hi + hi*lo + lo*hi) ----
        #pragma unroll
        for (int nt = 0; nt < 16; ++nt) {
            int n = nh * 256 + nt * 16 + ncol;
            const __bf16* bh = &aHi[n * A_STRIDE + kb];
            const __bf16* bl = &aLo[n * A_STRIDE + kb];
            bf16x8 h0 = *(const bf16x8*)(bh);
            bf16x8 h1 = *(const bf16x8*)(bh + 16);
            bf16x8 l0 = *(const bf16x8*)(bl);
            bf16x8 l1 = *(const bf16x8*)(bl + 16);
            bf16x16 bhi = __builtin_shufflevector(h0, h1, 0,1,2,3,4,5,6,7,8,9,10,11,12,13,14,15);
            bf16x16 blo = __builtin_shufflevector(l0, l1, 0,1,2,3,4,5,6,7,8,9,10,11,12,13,14,15);
            acc[nt] = __builtin_amdgcn_wmma_f32_16x16x32_bf16(
                false, ahi, false, bhi, (short)0, acc[nt], false, false);
            acc[nt] = __builtin_amdgcn_wmma_f32_16x16x32_bf16(
                false, ahi, false, blo, (short)0, acc[nt], false, false);
            acc[nt] = __builtin_amdgcn_wmma_f32_16x16x32_bf16(
                false, alo, false, bhi, (short)0, acc[nt], false, false);
        }
        __syncthreads();   // protect buffers before the next stage overwrites
    }

    // ======== fused epilogue ========
    // acc[nt][v]: row_local = mi*16 + lh*8 + v, col = nh*256 + nt*16 + ncol
    float psum[8] = {0.f,0.f,0.f,0.f,0.f,0.f,0.f,0.f};
    #pragma unroll
    for (int nt = 0; nt < 16; ++nt) {
        int gc = nh * 256 + nt * 16 + ncol;
        #pragma unroll
        for (int v = 0; v < 8; ++v) {
            int rl = mi * 16 + lh * 8 + v;
            float e   = EMIT[(size_t)(row0 + rl) * S_DIM + gc];
            float r   = acc[nt][v];
            float val = fmaxf(e, EPSV) * fmaxf(r, EPSV);
            acc[nt][v] = val;
            psum[v] += val;
        }
    }
    #pragma unroll
    for (int v = 0; v < 8; ++v) {
        float s = psum[v];
        s += __shfl_xor(s, 1, 32);
        s += __shfl_xor(s, 2, 32);
        s += __shfl_xor(s, 4, 32);
        s += __shfl_xor(s, 8, 32);
        psum[v] = s;
    }
    if (ncol == 0) {
        #pragma unroll
        for (int v = 0; v < 8; ++v)
            atomicAdd(&rowsum[mi * 16 + lh * 8 + v], psum[v]);   // ds_add_f32
    }
    __syncthreads();

    const size_t OUT1 = (size_t)Btot * (S_DIM + 1);
    const size_t OUT2 = OUT1 + (size_t)Btot * S_DIM;

    float inv[8], lgS[8];
    #pragma unroll
    for (int v = 0; v < 8; ++v) {
        float Ssum = rowsum[mi * 16 + lh * 8 + v];
        inv[v] = 1.0f / Ssum;
        lgS[v] = __logf(Ssum);
    }

    #pragma unroll
    for (int nt = 0; nt < 16; ++nt) {
        int gc = nh * 256 + nt * 16 + ncol;
        #pragma unroll
        for (int v = 0; v < 8; ++v) {
            int    rl   = mi * 16 + lh * 8 + v;
            size_t grow = (size_t)(row0 + rl);
            float  sfn  = acc[nt][v] * inv[v];
            out[OUT1 + grow * S_DIM + gc] = sfn;            // new_sf
            out[grow * (S_DIM + 1) + gc]  = __logf(sfn);    // log(new_sf)
        }
    }
    if (nh == 0 && ncol == 0) {
        #pragma unroll
        for (int v = 0; v < 8; ++v) {
            int    rl   = mi * 16 + lh * 8 + v;
            size_t grow = (size_t)(row0 + rl);
            float  nll  = LL[grow] + lgS[v];
            out[grow * (S_DIM + 1) + S_DIM] = nll;          // output[:, 512]
            out[OUT2 + grow]                = nll;          // new_ll
        }
    }
}

extern "C" void kernel_launch(void* const* d_in, const int* in_sizes, int n_in,
                              void* d_out, int out_size, void* d_ws, size_t ws_size,
                              hipStream_t stream) {
    const float* EMIT = (const float*)d_in[0];
    const float* SF   = (const float*)d_in[1];
    const float* LL   = (const float*)d_in[2];
    const float* A    = (const float*)d_in[3];
    float* out = (float*)d_out;

    int Btot = in_sizes[0] / S_DIM;          // 32768
    int grid = Btot / ROWS_PER_WG;           // 512

    const size_t need = (size_t)2 * S_DIM * S_DIM * sizeof(unsigned short);  // 1 MiB
    int useWs = (d_ws != nullptr && ws_size >= need) ? 1 : 0;

    unsigned short* AtHi = (unsigned short*)d_ws;
    unsigned short* AtLo = AtHi + (size_t)S_DIM * S_DIM;

    if (useWs)
        hipLaunchKernelGGL(prep_A_kernel, dim3(256), dim3(256), 0, stream, A, AtHi, AtLo);

    hipLaunchKernelGGL(hmm_fwd_step, dim3(grid), dim3(256), SMEM_TOTAL, stream,
                       EMIT, SF, LL, A, AtHi, AtLo, out, Btot, useWs);
}
